// IMPALA_12103217840818
// MI455X (gfx1250) — compile-verified, matrix operand order checked by reading, not run on hardware
//
#include <hip/hip_runtime.h>
#include <hip/hip_bf16.h>

// ---------------- problem constants ----------------
constexpr int S    = 512;
constexpr int B    = 256;
constexpr int AACT = 16;
constexpr int HID  = 64;
constexpr int R    = 256;
constexpr int IN   = HID + AACT + 1;   // 81
constexpr int K1   = 96;               // IN padded to multiple of 32
constexpr int G    = 4 * R;            // 1024 gate columns

constexpr int NBLK = 32;               // persistent blocks
constexpr int TPB  = 256;              // 8 waves per block -> 256 waves total

// packed weight tile counts
constexpr int KT_IH = K1 / 32;         // 3
constexpr int KT_HH = R  / 32;         // 8
constexpr int NT_G  = G  / 16;         // 64
constexpr int KT_HD = R  / 32;         // 8
constexpr int NT_HD = 2;               // 32 head cols = 16 actor + 1 critic + pad

// ---------------- vector types ----------------
typedef _Float16 v16h __attribute__((ext_vector_type(16)));
typedef _Float16 h8   __attribute__((ext_vector_type(8)));
typedef float    v8f  __attribute__((ext_vector_type(8)));

__device__ __forceinline__ v8f wmma16x16x32(v16h a, v16h b, v8f c) {
  return __builtin_amdgcn_wmma_f32_16x16x32_f16(false, a, false, b, (short)0, c,
                                                false, false);
}

// ISA 16-bit fragment half-index: halves 0..7 -> K = koff0 + {0..7},
// halves 8..15 -> K = 16 + koff0 + {0..7}, koff0 = (lane>=16) ? 8 : 0
__device__ __host__ __forceinline__ int klocal_of(int lane, int j) {
  int k0 = (lane & 16) ? 8 : 0;
  return (j < 8) ? (k0 + j) : (16 + k0 + (j - 8));
}

// A fragment (16x32 f16) from row-major f16 matrix
__device__ __forceinline__ v16h load_afrag(const _Float16* __restrict__ src,
                                           int ld, int row0, int k0, int lane) {
  int r = row0 + (lane & 15);
  int kb = k0 + ((lane & 16) ? 8 : 0);
  const h8* p0 = reinterpret_cast<const h8*>(src + (size_t)r * ld + kb);
  const h8* p1 = reinterpret_cast<const h8*>(src + (size_t)r * ld + kb + 16);
  h8 lo = *p0, hi = *p1;
  return __builtin_shufflevector(lo, hi, 0, 1, 2, 3, 4, 5, 6, 7,
                                 8, 9, 10, 11, 12, 13, 14, 15);
}

// same, but non-temporal (X tensor is streamed exactly once)
__device__ __forceinline__ v16h load_afrag_nt(const _Float16* __restrict__ src,
                                              int ld, int row0, int k0,
                                              int lane) {
  int r = row0 + (lane & 15);
  int kb = k0 + ((lane & 16) ? 8 : 0);
  const h8* p0 = reinterpret_cast<const h8*>(src + (size_t)r * ld + kb);
  const h8* p1 = reinterpret_cast<const h8*>(src + (size_t)r * ld + kb + 16);
  h8 lo = __builtin_nontemporal_load(p0);
  h8 hi = __builtin_nontemporal_load(p1);
  return __builtin_shufflevector(lo, hi, 0, 1, 2, 3, 4, 5, 6, 7,
                                 8, 9, 10, 11, 12, 13, 14, 15);
}

// B fragment from pre-packed layout: one contiguous 32B chunk per lane
__device__ __forceinline__ v16h load_bfrag(const _Float16* __restrict__ packed,
                                           int tile, int lane) {
  return *reinterpret_cast<const v16h*>(packed + ((size_t)tile * 32 + lane) * 16);
}

__device__ __forceinline__ float sigm(float x) { return 1.f / (1.f + __expf(-x)); }

__device__ __forceinline__ v8f splat8(float x) {
  v8f v = {x, x, x, x, x, x, x, x};
  return v;
}

// ---------------- prep kernel ----------------
__global__ __launch_bounds__(TPB) void impala_prep(
    const float* __restrict__ input, const int* __restrict__ last_action,
    const float* __restrict__ reward, const float* __restrict__ hidden,
    const float* __restrict__ fc_w, const float* __restrict__ fc_b,
    const float* __restrict__ w_ih, const float* __restrict__ w_hh,
    const float* __restrict__ b_ih, const float* __restrict__ b_hh,
    const float* __restrict__ actor_w, const float* __restrict__ critic_w,
    _Float16* __restrict__ X, _Float16* __restrict__ Wih_p,
    _Float16* __restrict__ Whh_p, _Float16* __restrict__ Whead_p,
    float* __restrict__ bias, _Float16* __restrict__ hbuf0,
    int* __restrict__ bar) {
  const int tid = blockIdx.x * TPB + threadIdx.x;
  const int NT  = gridDim.x * TPB;  // == S*B

  // --- X row: fc + reward + onehot + pad ---
  if (tid < S * B) {
    const float* inp = input + (size_t)tid * 16;
    _Float16* xr = X + (size_t)tid * K1;
    float in_reg[16];
#pragma unroll
    for (int k = 0; k < 16; ++k) in_reg[k] = inp[k];
    for (int c = 0; c < HID; ++c) {
      float s = fc_b[c];
#pragma unroll
      for (int k = 0; k < 16; ++k) s += in_reg[k] * fc_w[c * 16 + k];
      xr[c] = (_Float16)s;
    }
    xr[HID] = (_Float16)reward[tid];
    int act = last_action[tid];
#pragma unroll
    for (int a = 0; a < AACT; ++a)
      xr[HID + 1 + a] = (_Float16)((a == act) ? 1.f : 0.f);
#pragma unroll
    for (int p = IN; p < K1; ++p) xr[p] = (_Float16)0.f;
  }

  // --- pack W_ih (K1 x G), zero-padded beyond k>=IN ---
  for (int i = tid; i < KT_IH * NT_G * 32 * 16; i += NT) {
    int j = i & 15, l = (i >> 4) & 31, nt = (i >> 9) & 63, kt = i >> 15;
    int k = kt * 32 + klocal_of(l, j);
    int col = nt * 16 + (l & 15);
    float v = (k < IN) ? w_ih[(size_t)col * IN + k] : 0.f;
    Wih_p[i] = (_Float16)v;
  }

  // --- pack W_hh (R x G) ---
  for (int i = tid; i < KT_HH * NT_G * 32 * 16; i += NT) {
    int j = i & 15, l = (i >> 4) & 31, nt = (i >> 9) & 63, kt = i >> 15;
    int k = kt * 32 + klocal_of(l, j);
    int col = nt * 16 + (l & 15);
    Whh_p[i] = (_Float16)w_hh[(size_t)col * R + k];
  }

  // --- pack heads (R x 32): cols 0..15 actor, 16 critic, rest 0 ---
  for (int i = tid; i < KT_HD * NT_HD * 32 * 16; i += NT) {
    int j = i & 15, l = (i >> 4) & 31, nt = (i >> 9) & 1, kt = i >> 10;
    int k = kt * 32 + klocal_of(l, j);
    int col = nt * 16 + (l & 15);
    float v = 0.f;
    if (col < AACT) v = actor_w[(size_t)col * R + k];
    else if (col == AACT) v = critic_w[k];
    Whead_p[i] = (_Float16)v;
  }

  // --- fused gate bias ---
  for (int i = tid; i < G; i += NT) bias[i] = b_ih[i] + b_hh[i];

  // --- initial h (f16, buffer 0); c is loaded to registers by lstm kernel ---
  for (int i = tid; i < B * R; i += NT) hbuf0[i] = (_Float16)hidden[i];

  // --- barrier state ---
  if (tid == 0) { bar[0] = 0; bar[1] = 0; }
}

// ---------------- grid barrier ----------------
__device__ __forceinline__ void grid_barrier(int* counter, int* gen) {
  __threadfence();
  __syncthreads();
  if (threadIdx.x == 0) {
    int g = __hip_atomic_load(gen, __ATOMIC_RELAXED, __HIP_MEMORY_SCOPE_AGENT);
    int v = __hip_atomic_fetch_add(counter, 1, __ATOMIC_ACQ_REL,
                                   __HIP_MEMORY_SCOPE_AGENT);
    if (v == NBLK - 1) {
      __hip_atomic_store(counter, 0, __ATOMIC_RELAXED, __HIP_MEMORY_SCOPE_AGENT);
      __hip_atomic_fetch_add(gen, 1, __ATOMIC_RELEASE, __HIP_MEMORY_SCOPE_AGENT);
    } else {
      while (__hip_atomic_load(gen, __ATOMIC_ACQUIRE,
                               __HIP_MEMORY_SCOPE_AGENT) == g) {
        __builtin_amdgcn_s_sleep(2);
      }
    }
  }
  __syncthreads();
  __threadfence();
}

// ---------------- persistent LSTM kernel ----------------
// 256 waves, one (batch-tile m, r-tile n) block per wave, c held in VGPRs,
// weight fragments held in VGPRs. Per step:
//   X-GEMM (pre-barrier) | barrier | h A-frag clause -> heads(t-1) -> h-GEMM.
__global__ __launch_bounds__(TPB) void impala_lstm(
    const unsigned char* __restrict__ done, const float* __restrict__ hidden,
    const float* __restrict__ actor_b, const float* __restrict__ critic_b,
    const _Float16* __restrict__ X, const _Float16* __restrict__ Wih_p,
    const _Float16* __restrict__ Whh_p, const _Float16* __restrict__ Whead_p,
    const float* __restrict__ bias, _Float16* __restrict__ hbuf0,
    _Float16* __restrict__ hbuf1, float* __restrict__ out_logits,
    float* __restrict__ out_values, int* __restrict__ bar) {
  const int lane = threadIdx.x & 31;
  const int w    = blockIdx.x * (TPB >> 5) + (threadIdx.x >> 5);  // 0..255
  const int m = w >> 4;    // batch tile 0..15
  const int n = w & 15;    // r tile     0..15
  int* cnt = bar;
  int* gen = bar + 1;

  const v16h hzero = {};

  // ---- preload weight B-fragments into registers (loop-invariant) ----
  v16h wih_f[KT_IH][4];
#pragma unroll
  for (int kt = 0; kt < KT_IH; ++kt)
#pragma unroll
    for (int q = 0; q < 4; ++q)
      wih_f[kt][q] = load_bfrag(Wih_p, kt * NT_G + (q * 16 + n), lane);

  v16h whh_f[KT_HH][4];
#pragma unroll
  for (int kt = 0; kt < KT_HH; ++kt)
#pragma unroll
    for (int q = 0; q < 4; ++q)
      whh_f[kt][q] = load_bfrag(Whh_p, kt * NT_G + (q * 16 + n), lane);

  // head fragments (only waves 0..31 use them)
  const int mt = w >> 1, nt = w & 1;
  v16h whd_f[KT_HD];
  float head_bias = 0.f;
  if (w < 32) {
#pragma unroll
    for (int kt = 0; kt < KT_HD; ++kt)
      whd_f[kt] = load_bfrag(Whead_p, kt * NT_HD + nt, lane);
    head_bias = (nt == 0) ? actor_b[lane & 15] : critic_b[0];
  }

  // ---- loop-invariant scalars ----
  const int col = n * 16 + (lane & 15);            // r index 0..255
  const float bI = bias[0 * R + col];
  const float bF = bias[1 * R + col];
  const float bG = bias[2 * R + col];
  const float bO = bias[3 * R + col];
  const int rowbase = m * 16 + ((lane & 16) ? 8 : 0);
  const int arow    = m * 16 + (lane & 15);        // A-fragment row for this lane

  // ---- c block lives in registers for the whole sequence ----
  v8f creg;
  {
    const float* c0 = hidden + (size_t)B * R;
#pragma unroll
    for (int j = 0; j < 8; ++j) creg[j] = c0[(size_t)(rowbase + j) * R + col];
  }

  auto do_heads = [&](int tprev, const _Float16* __restrict__ hsrc) {
    // load all A-fragments first (one clause), then dense WMMA run
    v16h a[KT_HD];
#pragma unroll
    for (int kt = 0; kt < KT_HD; ++kt)
      a[kt] = load_afrag(hsrc, R, mt * 16, kt * 32, lane);
    v8f acc = splat8(head_bias);
#pragma unroll
    for (int kt = 0; kt < KT_HD; ++kt) acc = wmma16x16x32(a[kt], whd_f[kt], acc);
    const int colN = lane & 15;
    const int rb = mt * 16 + ((lane & 16) ? 8 : 0);
    if (nt == 0) {
#pragma unroll
      for (int j = 0; j < 8; ++j)
        out_logits[((size_t)tprev * B + rb + j) * AACT + colN] = acc[j];
    } else if (colN == 0) {
#pragma unroll
      for (int j = 0; j < 8; ++j)
        out_values[(size_t)tprev * B + rb + j] = acc[j];
    }
  };

  for (int t = 0; t < S; ++t) {
    // ---- X-part GEMM: independent of other waves' step t-1 results ----
    v8f acc[4];
    acc[0] = splat8(bI);
    acc[1] = splat8(bF);
    acc[2] = splat8(bG);
    acc[3] = splat8(bO);
    {
      const _Float16* Xt = X + (size_t)t * B * K1;
      v16h xa[KT_IH];
#pragma unroll
      for (int kt = 0; kt < KT_IH; ++kt)
        xa[kt] = load_afrag_nt(Xt, K1, m * 16, kt * 32, lane);
#pragma unroll
      for (int kt = 0; kt < KT_IH; ++kt)
#pragma unroll
        for (int q = 0; q < 4; ++q)
          acc[q] = wmma16x16x32(xa[kt], wih_f[kt][q], acc[q]);
    }

    // ---- wait for step t-1's h (written to hbuf[t&1]) ----
    grid_barrier(cnt, gen);

    const _Float16* hr = (t & 1) ? hbuf1 : hbuf0;
    _Float16*       hw = (t & 1) ? hbuf0 : hbuf1;
    const unsigned char* dt_ = done + (size_t)t * B;

    // ---- issue ALL recurrent A-fragment loads first (one clause) ----
    v16h ha[KT_HH];
#pragma unroll
    for (int kt = 0; kt < KT_HH; ++kt)
      ha[kt] = load_afrag(hr, R, m * 16, kt * 32, lane);

    // per-row done flags (used for both A-row masking and c masking)
    const bool arow_done = dt_[arow] != 0;
    bool cdone[8];
#pragma unroll
    for (int j = 0; j < 8; ++j) cdone[j] = dt_[rowbase + j] != 0;

    // ---- heads for step t-1: overlaps the in-flight h loads ----
    if (w < 32 && t > 0) do_heads(t - 1, hr);

    // ---- recurrent GEMM, dense WMMA run with in-register done masking ----
#pragma unroll
    for (int kt = 0; kt < KT_HH; ++kt) {
      v16h a = arow_done ? hzero : ha[kt];
#pragma unroll
      for (int q = 0; q < 4; ++q)
        acc[q] = wmma16x16x32(a, whh_f[kt][q], acc[q]);
    }

    // ---- LSTM pointwise update; c stays in registers ----
#pragma unroll
    for (int j = 0; j < 8; ++j) {
      float c_old = cdone[j] ? 0.f : creg[j];
      float iv = sigm(acc[0][j]);
      float fv = sigm(acc[1][j]);
      float gv = tanhf(acc[2][j]);
      float ov = sigm(acc[3][j]);
      float cn = fv * c_old + iv * gv;
      float hn = ov * tanhf(cn);
      creg[j] = cn;
      hw[(size_t)(rowbase + j) * R + col] = (_Float16)hn;
    }
    // hbuf[(t+1)&1] (= hw) is only read after the next barrier -> safe.
  }

  // ---- final heads for step S-1 ----
  grid_barrier(cnt, gen);
  if (w < 32) do_heads(S - 1, (S & 1) ? hbuf1 : hbuf0);
}

// ---------------- launcher ----------------
extern "C" void kernel_launch(void* const* d_in, const int* in_sizes, int n_in,
                              void* d_out, int out_size, void* d_ws,
                              size_t ws_size, hipStream_t stream) {
  (void)in_sizes; (void)n_in; (void)out_size; (void)ws_size;

  const float* input       = (const float*)d_in[0];
  const int*   last_action = (const int*)d_in[1];
  const float* reward      = (const float*)d_in[2];
  const unsigned char* done = (const unsigned char*)d_in[3];
  const float* hidden      = (const float*)d_in[4];
  const float* fc_w        = (const float*)d_in[5];
  const float* fc_b        = (const float*)d_in[6];
  const float* w_ih        = (const float*)d_in[7];
  const float* w_hh        = (const float*)d_in[8];
  const float* b_ih        = (const float*)d_in[9];
  const float* b_hh        = (const float*)d_in[10];
  const float* actor_w     = (const float*)d_in[11];
  const float* actor_b     = (const float*)d_in[12];
  const float* critic_w    = (const float*)d_in[13];
  const float* critic_b    = (const float*)d_in[14];

  char* ws = (char*)d_ws;
  size_t off = 0;
  auto take = [&](size_t bytes) {
    void* p = ws + off;
    off += (bytes + 255) & ~(size_t)255;
    return p;
  };
  _Float16* X       = (_Float16*)take((size_t)S * B * K1 * 2);
  _Float16* Wih_p   = (_Float16*)take((size_t)KT_IH * NT_G * 32 * 16 * 2);
  _Float16* Whh_p   = (_Float16*)take((size_t)KT_HH * NT_G * 32 * 16 * 2);
  _Float16* Whead_p = (_Float16*)take((size_t)KT_HD * NT_HD * 32 * 16 * 2);
  float*    bias    = (float*)take((size_t)G * 4);
  _Float16* hbuf0   = (_Float16*)take((size_t)B * R * 2);
  _Float16* hbuf1   = (_Float16*)take((size_t)B * R * 2);
  int*      bar     = (int*)take(256);

  float* out_logits = (float*)d_out;
  float* out_values = out_logits + (size_t)S * B * AACT;

  impala_prep<<<(S * B) / TPB, TPB, 0, stream>>>(
      input, last_action, reward, hidden, fc_w, fc_b, w_ih, w_hh, b_ih, b_hh,
      actor_w, critic_w, X, Wih_p, Whh_p, Whead_p, bias, hbuf0, bar);

  impala_lstm<<<NBLK, TPB, 0, stream>>>(
      done, hidden, actor_b, critic_b, X, Wih_p, Whh_p, Whead_p, bias, hbuf0,
      hbuf1, out_logits, out_values, bar);
}
